// MultiheadAttention_11244224381448
// MI455X (gfx1250) — matrix operand order repaired and checked by reading, not compiled, not run on hardware
//
#include <hip/hip_runtime.h>
#include <math.h>

typedef __attribute__((ext_vector_type(16))) _Float16 v16h;
typedef __attribute__((ext_vector_type(8)))  _Float16 v8h;
typedef __attribute__((ext_vector_type(8)))  float    v8f;
typedef __attribute__((ext_vector_type(4)))  float    v4f;

#define H_ 12
#define D_ 64
#define B_ 4
#define T_ 2048
#define P_ (H_*D_)       /* 768 */
#define BT_ (B_*T_)      /* 8192 */
#define SCALE_ 0.125f    /* 1/sqrt(64) */

union V16 { v16h v; v8h h[2]; };

static __device__ __forceinline__ v8f wmma16(v16h a, v16h b, v8f c) {
  // D = A(16x32 f16) * B(32x16 f16) + C(16x16 f32)
  return __builtin_amdgcn_wmma_f32_16x16x32_f16(false, a, false, b, (short)0, c,
                                                false, false);
}

// ---------------------------------------------------------------------------
// Kernel 1: QKV projection.  out = hidden[8192x64] @ W[64x768] + bias,
// 16x16 tiles, one wave per tile, 2 WMMAs (K=64).
// mat = blockIdx.y (0=Q,1=K,2=V) so all mat branches are scalar-uniform.
// Q,K stored f16 [B,H,T,D]; V stored f16 transposed [B,H,D,T].
// ---------------------------------------------------------------------------
__global__ __launch_bounds__(256) void qkv_proj(
    const float* __restrict__ hs,
    const float* __restrict__ Wq, const float* __restrict__ bq,
    const float* __restrict__ Wk, const float* __restrict__ bk,
    const float* __restrict__ Wv, const float* __restrict__ bv,
    _Float16* __restrict__ qd, _Float16* __restrict__ kd,
    _Float16* __restrict__ vtd)
{
  const int wave = blockIdx.x * 8 + (threadIdx.x >> 5);
  const int lane = threadIdx.x & 31;
  const int ln = lane & 15;        // column/row-within-group
  const int hh = lane >> 4;        // lane half select

  const int mat = blockIdx.y;                 // scalar-uniform
  const int nT  = wave % (P_/16);
  const int mT  = wave / (P_/16);
  const int m0 = mT * 16, n0 = nT * 16;

  const float* W;  const float* bias;
  if (mat == 0)      { W = Wq; bias = bq; }
  else if (mat == 1) { W = Wk; bias = bk; }
  else               { W = Wv; bias = bv; }

  // ---- A operands: hidden rows, converted f32 -> f16, ISA A-layout ----
  const float* arow = hs + (size_t)(m0 + ln) * D_;
  V16 a[2];
#pragma unroll
  for (int kk = 0; kk < 2; ++kk) {
    const int k0 = kk * 32;
    v4f f0 = *(const v4f*)(arow + k0 + 8*hh);
    v4f f1 = *(const v4f*)(arow + k0 + 8*hh + 4);
    v4f f2 = *(const v4f*)(arow + k0 + 16 + 8*hh);
    v4f f3 = *(const v4f*)(arow + k0 + 16 + 8*hh + 4);
#pragma unroll
    for (int j = 0; j < 4; ++j) {
      a[kk].v[j]      = (_Float16)f0[j];
      a[kk].v[4 + j]  = (_Float16)f1[j];
      a[kk].v[8 + j]  = (_Float16)f2[j];
      a[kk].v[12 + j] = (_Float16)f3[j];
    }
  }

  // ---- B operands: W columns, ISA B-layout (lane = N, halves = K) ----
  const int n = n0 + ln;
  V16 bm[2];
#pragma unroll
  for (int kk = 0; kk < 2; ++kk) {
    const int dbase = kk * 32 + 8 * hh;
#pragma unroll
    for (int j = 0; j < 16; ++j)
      bm[kk].v[j] = (_Float16)W[(size_t)(dbase + j + (j >= 8 ? 8 : 0)) * P_ + n];
  }

  // ---- bias in C, two chained WMMAs over K=64 ----
  const float bval = bias[n];
  v8f c;
#pragma unroll
  for (int r = 0; r < 8; ++r) c[r] = bval;
  c = wmma16(a[0].v, bm[0].v, c);
  c = wmma16(a[1].v, bm[1].v, c);

  // ---- scatter to f16 workspace (mat branch hoisted: scalar-uniform) ----
  const int col = n0 + ln;
  const int hd = col >> 6, d = col & 63;
  const int row0 = m0 + 8*hh;                 // C layout: M = r + 8*(lane>=16)
  const int bb = row0 >> 11;                  // tile never straddles batches
  const int t0 = row0 & (T_ - 1);
  const size_t base = (size_t)(bb * H_ + hd);

  if (mat == 2) {
    // V transposed: lane's 8 rows are consecutive t -> one b128 store
    v8h pack;
#pragma unroll
    for (int r = 0; r < 8; ++r) pack[r] = (_Float16)c[r];
    *(v8h*)(vtd + (base * D_ + d) * T_ + t0) = pack;
  } else if (mat == 0) {
#pragma unroll
    for (int r = 0; r < 8; ++r)
      qd[(base * T_ + (t0 + r)) * D_ + d] = (_Float16)c[r];
  } else {
#pragma unroll
    for (int r = 0; r < 8; ++r)
      kd[(base * T_ + (t0 + r)) * D_ + d] = (_Float16)c[r];
  }
}

// ---------------------------------------------------------------------------
// Kernel 2: flash attention.  One wave owns a 16-query tile of one (b,h);
// loops over keys in chunks of 32 with online softmax.
// Per chunk: 4 WMMAs for S = Q*K^T (two 16-key subtiles x K=64),
//            4 WMMAs for acc = P*V + alpha*acc (4 d-tiles, K=32).
// ---------------------------------------------------------------------------
__global__ __launch_bounds__(256) void flash_attn(
    const _Float16* __restrict__ qd, const _Float16* __restrict__ kd,
    const _Float16* __restrict__ vtd, const float* __restrict__ mask,
    float* __restrict__ out)
{
  __shared__ _Float16 pbuf[8][16][32];     // per-wave P re-layout buffer (8 KB)

  const int wslot = threadIdx.x >> 5;
  const int wave  = blockIdx.x * 8 + wslot;
  const int lane  = threadIdx.x & 31;
  const int ln = lane & 15;
  const int hh = lane >> 4;

  const int qT = wave & (T_/16 - 1);       // 128 query tiles per (b,h)
  const int bh = wave >> 7;                // 0..47
  const int bb = bh / H_, hd = bh % H_;
  const int q0 = qT * 16;

  // ---- Q A-operands (held in registers for the whole loop) ----
  const _Float16* qrow = qd + ((size_t)bh * T_ + q0 + ln) * D_;
  V16 aQ[2];
#pragma unroll
  for (int kk = 0; kk < 2; ++kk) {
    aQ[kk].h[0] = *(const v8h*)(qrow + kk*32 + 16*hh);
    aQ[kk].h[1] = *(const v8h*)(qrow + kk*32 + 16*hh + 8);
  }

  v8f acc0 = {}, acc1 = {}, acc2 = {}, acc3 = {};
  float mrow[8], lrow[8];
#pragma unroll
  for (int r = 0; r < 8; ++r) { mrow[r] = -1e30f; lrow[r] = 0.f; }

  const float* mrk = mask + (size_t)bb * T_;
  const _Float16* kbase = kd + (size_t)bh * T_ * D_;
  const _Float16* vbase = vtd + (size_t)bh * D_ * T_;

  for (int j0 = 0; j0 < T_; j0 += 32) {
    // prefetch next chunk's K rows / V columns (global_prefetch_b8)
    if (j0 + 32 < T_) {
      __builtin_prefetch(kbase + (size_t)(j0 + 32 + ln) * D_ + 16*hh, 0, 1);
      __builtin_prefetch(vbase + (size_t)(ln) * T_ + j0 + 32 + 16*hh, 0, 1);
    }

    // ---- scores: two 16-key subtiles, K=64 each ----
    v8f s[2];
#pragma unroll
    for (int sb = 0; sb < 2; ++sb) {
      const _Float16* krow = kbase + (size_t)(j0 + sb*16 + ln) * D_;
      V16 b0, b1;
      b0.h[0] = *(const v8h*)(krow + 16*hh);
      b0.h[1] = *(const v8h*)(krow + 16*hh + 8);
      b1.h[0] = *(const v8h*)(krow + 32 + 16*hh);
      b1.h[1] = *(const v8h*)(krow + 32 + 16*hh + 8);
      v8f z = {};
      z = wmma16(aQ[0].v, b0.v, z);
      z = wmma16(aQ[1].v, b1.v, z);
      s[sb] = z;
    }

    const float mk0 = (1.f - mrk[j0 + ln])      * -10000.f;
    const float mk1 = (1.f - mrk[j0 + 16 + ln]) * -10000.f;

    float alpha[8];
#pragma unroll
    for (int r = 0; r < 8; ++r) {
      float s0 = s[0][r] * SCALE_ + mk0;
      float s1 = s[1][r] * SCALE_ + mk1;
      // row max across the 16 lanes holding this row
      float rm = fmaxf(s0, s1);
#pragma unroll
      for (int off = 1; off < 16; off <<= 1)
        rm = fmaxf(rm, __shfl_xor(rm, off, 16));
      const float mnew = fmaxf(mrow[r], rm);
      const float al = __expf(mrow[r] - mnew);
      const float p0 = __expf(s0 - mnew);
      const float p1 = __expf(s1 - mnew);
      float rs = p0 + p1;
#pragma unroll
      for (int off = 1; off < 16; off <<= 1)
        rs += __shfl_xor(rs, off, 16);
      lrow[r] = lrow[r] * al + rs;
      mrow[r] = mnew;
      alpha[r] = al;
      // stash P (C layout -> (M,key) matrix in LDS)
      const int M = r + 8*hh;
      pbuf[wslot][M][ln]      = (_Float16)p0;
      pbuf[wslot][M][16 + ln] = (_Float16)p1;
    }

    // per-wave LDS RAW: DS ops are in-order within a wave; barrier the
    // compiler and drain DScnt before re-reading.
    asm volatile("s_wait_dscnt 0" ::: "memory");

    // ---- reload P in A-layout ----
    V16 aP;
    aP.h[0] = *(const v8h*)(&pbuf[wslot][ln][16*hh]);
    aP.h[1] = *(const v8h*)(&pbuf[wslot][ln][16*hh + 8]);

    // ---- rescale accumulators by alpha (per row), then P @ V ----
#pragma unroll
    for (int r = 0; r < 8; ++r) {
      acc0[r] *= alpha[r]; acc1[r] *= alpha[r];
      acc2[r] *= alpha[r]; acc3[r] *= alpha[r];
    }
#pragma unroll
    for (int dt = 0; dt < 4; ++dt) {
      const _Float16* vrow = vbase + (size_t)(dt*16 + ln) * T_ + j0;
      V16 bV;
      bV.h[0] = *(const v8h*)(vrow + 16*hh);
      bV.h[1] = *(const v8h*)(vrow + 16*hh + 8);
      v8f& a = (dt == 0) ? acc0 : (dt == 1) ? acc1 : (dt == 2) ? acc2 : acc3;
      a = wmma16(aP.v, bV.v, a);
    }
  }

  // ---- normalize and write fp32 output [B, T, H*D] ----
#pragma unroll
  for (int r = 0; r < 8; ++r) {
    const int t = q0 + r + 8*hh;
    const float inv = 1.f / lrow[r];
    float* orow = out + ((size_t)bb * T_ + t) * P_ + hd * D_;
    orow[0*16 + ln] = acc0[r] * inv;
    orow[1*16 + ln] = acc1[r] * inv;
    orow[2*16 + ln] = acc2[r] * inv;
    orow[3*16 + ln] = acc3[r] * inv;
  }
}

// ---------------------------------------------------------------------------
extern "C" void kernel_launch(void* const* d_in, const int* in_sizes, int n_in,
                              void* d_out, int out_size, void* d_ws, size_t ws_size,
                              hipStream_t stream) {
  const float* hs   = (const float*)d_in[0];
  const float* mask = (const float*)d_in[1];
  const float* Wq = (const float*)d_in[2];
  const float* bq = (const float*)d_in[3];
  const float* Wk = (const float*)d_in[4];
  const float* bk = (const float*)d_in[5];
  const float* Wv = (const float*)d_in[6];
  const float* bv = (const float*)d_in[7];
  float* out = (float*)d_out;

  const size_t N = (size_t)B_ * H_ * T_ * D_;   // 6.29M halves each
  _Float16* qd  = (_Float16*)d_ws;
  _Float16* kd  = qd + N;
  _Float16* vtd = kd + N;

  // Kernel 1: (8192/16)*(768/16) = 24576 tiles per matrix, 8 waves/block,
  // grid.y selects Q/K/V so the branch is scalar-uniform.
  dim3 g1((BT_/16) * (P_/16) / 8, 3, 1);
  qkv_proj<<<g1, 256, 0, stream>>>(hs, Wq, bq, Wk, bk, Wv, bv, qd, kd, vtd);

  // Kernel 2: B*H*(T/16) = 6144 waves, 8 waves/block
  const int waves2 = B_ * H_ * (T_/16);
  flash_attn<<<waves2 / 8, 256, 0, stream>>>(qd, kd, vtd, mask, out);
}
